// AbsDiagNetGated_10462540333633
// MI455X (gfx1250) — matrix operand order repaired and checked by reading, not compiled, not run on hardware
//
#include <hip/hip_runtime.h>

#define DEVFN __device__ __forceinline__

typedef __bf16 bf16_t;
typedef __attribute__((ext_vector_type(16))) __bf16 bf16x16;
typedef __attribute__((ext_vector_type(8)))  __bf16 bf16x8;
typedef __attribute__((ext_vector_type(8)))  float  floatx8;
typedef __attribute__((ext_vector_type(4)))  unsigned int u32x4;
typedef __attribute__((ext_vector_type(8)))  int i32x8;
typedef __attribute__((ext_vector_type(4)))  int i32x4;

constexpr int T_ = 512, B_ = 128, I_ = 512, G_ = 1024, H_ = 1024, O_ = 512;
constexpr int TC_ = 64;              // timestep chunk
constexpr int MC_ = TC_ * B_;        // 8192 rows per chunk

// LDS tile geometry: 128 rows x 32 k-elements, padded to 40 elements (80 B)
// per row so 16-lane fragment reads stride 20 banks -> conflict-free.
// The TDM reproduces this padding: every 16 DWORDs stored, skip 4 DWORDs.
constexpr int LDS_LD = 40;           // elements per LDS tile row

DEVFN unsigned short f32_to_bf16_rne(float x) {
  unsigned int u = __float_as_uint(x);
  unsigned int r = u + 0x7FFFu + ((u >> 16) & 1u);
  return (unsigned short)(r >> 16);
}

union FragU { bf16x16 v; bf16x8 h[2]; };

DEVFN unsigned lds_off_u32(const void* p) {
  // generic LDS address: addr[31:0] is the wave-relative LDS offset
  return (unsigned)(size_t)p;
}

// ---- CDNA5 Tensor Data Mover: DMA one 128x32 bf16 tile into LDS ----------
// D# per CDNA5 ISA: group0 = {flags, lds_addr, global_addr,
// type=2}; group1 = {data_size=2B, pad 16DW->4DW, dims, tile dims, stride}.
DEVFN void tdm_load_tile(unsigned lds_byte_off, const void* gaddr,
                         int k_len, long row_stride_elems) {
  const unsigned long long ga = (unsigned long long)gaddr;
  u32x4 g0;
  g0[0] = 1u;                                        // count=1, load, no gather
  g0[1] = lds_byte_off;                              // lds_addr [63:32]
  g0[2] = (unsigned)(ga & 0xFFFFFFFFu);              // global_addr [95:64]
  g0[3] = (unsigned)((ga >> 32) & 0x01FFFFFFu)       // global_addr [120:96]
          | (2u << 30);                              // type=2 ("image")
  i32x8 g1;
  g1[0] = (1 << 16)                                  // data_size = 2 bytes
        | (1 << 20)                                  // pad_enable
        | (3 << 22)                                  // pad_interval: 16 DWORDs
        | (3 << 25);                                 // pad_amount: 4 DWORDs
  g1[1] = (k_len & 0xFFFF) << 16;                    // tensor_dim0[15:0]
  g1[2] = ((k_len >> 16) & 0xFFFF)                   // tensor_dim0[31:16]
        | (128 << 16);                               // tensor_dim1[15:0] = 128
  g1[3] = (32 << 16);                                // tile_dim0 = 32 elems
  g1[4] = 128;                                       // tile_dim1 = 128, tile_dim2=0
  g1[5] = (int)(row_stride_elems & 0xFFFFFFFF);      // tensor_dim0_stride[31:0]
  g1[6] = (int)((row_stride_elems >> 32) & 0xFFFF);  // stride[47:32]; dim1_stride=0
  g1[7] = 0;
  // groups 2/3: unused dims, kept benign (tensor_dim2=1, tile_dim3/4=0)
  i32x4 g2; g2[0] = 1; g2[1] = 1; g2[2] = 0; g2[3] = 0;
  i32x4 g3; g3[0] = 0; g3[1] = (1 << 16); g3[2] = 0; g3[3] = 0;
#if __clang_major__ >= 23
  i32x8 z8 = {0, 0, 0, 0, 0, 0, 0, 0};
  __builtin_amdgcn_tensor_load_to_lds(g0, g1, g2, g3, z8, 0);
#else
  __builtin_amdgcn_tensor_load_to_lds(g0, g1, g2, g3, 0);
#endif
}

// ---- WMMA fragment loads from an LDS tile (row stride LDS_LD) -----------
// A fragment 16x32: lane L(0..15) row m0+L, K {0..7,16..23}; lane L+16: {8..15,24..31}
DEVFN bf16x16 lds_frag_a(const unsigned short* tile, int m0) {
  const int lane = threadIdx.x & 31;
  const unsigned short* p =
      tile + (m0 + (lane & 15)) * LDS_LD + ((lane >> 4) << 3);
  FragU f;
  f.h[0] = *(const bf16x8*)(p);        // 16B aligned
  f.h[1] = *(const bf16x8*)(p + 16);   // 16B aligned
  return f.v;
}

// B fragment 32x16 with B[k][n] = W[n][k]: lane L col n0+L, K {0..7,16..23};
// lane L+16: K {8..15,24..31}.
DEVFN bf16x16 lds_frag_b(const unsigned short* tile, int n0) {
  const int lane = threadIdx.x & 31;
  const unsigned short* p =
      tile + (n0 + (lane & 15)) * LDS_LD + ((lane >> 4) << 3);
  FragU f;
  f.h[0] = *(const bf16x8*)(p);
  f.h[1] = *(const bf16x8*)(p + 16);
  return f.v;
}

// C = A * W^T.  MODE 0: store fp32.  MODE 1: relu + store bf16.
// Block: 256 threads = 8 waves as 4(M) x 2(N); wave tile 32x64;
// block tile 128x128; K staged by the TDM through double-buffered LDS
// in steps of 32, synchronized with s_wait_tensorcnt + barrier.
template <int MODE>
__launch_bounds__(256)
__global__ void wmma_gemm_tdm(const bf16_t* __restrict__ A,
                              const bf16_t* __restrict__ W,
                              float* __restrict__ Cf,
                              unsigned short* __restrict__ Cb,
                              int M, int N, int K) {
  __shared__ __align__(32) unsigned short Abuf[2][128 * LDS_LD];
  __shared__ __align__(32) unsigned short Bbuf[2][128 * LDS_LD];

  const int wave = threadIdx.x >> 5;
  const int lane = threadIdx.x & 31;
  const int blockM = blockIdx.y * 128;
  const int blockN = blockIdx.x * 128;
  const int wm = (wave >> 1) * 32;  // 0,32,64,96
  const int wn = (wave & 1) * 64;   // 0,64

  const bf16_t* Abase = A + (size_t)blockM * K;
  const bf16_t* Wbase = W + (size_t)blockN * K;

  floatx8 acc[2][4];
#pragma unroll
  for (int i = 0; i < 2; ++i)
#pragma unroll
    for (int j = 0; j < 4; ++j)
#pragma unroll
      for (int r = 0; r < 8; ++r) acc[i][j][r] = 0.f;

  // prologue: wave 0 DMAs k-slice 0 (one TDM op per tile)
  if (wave == 0) {
    tdm_load_tile(lds_off_u32(Abuf[0]), Abase, K, K);
    tdm_load_tile(lds_off_u32(Bbuf[0]), Wbase, K, K);
  }

  const int nst = K >> 5;
  for (int s = 0; s < nst; ++s) {
    const int cur = s & 1;
    __builtin_amdgcn_s_wait_tensorcnt(0);  // wave0: tiles landed; others: no-op
    __syncthreads();                       // publish; prior buf[cur^1] reads done
    if (s + 1 < nst && wave == 0) {        // overlap next DMA with WMMAs
      tdm_load_tile(lds_off_u32(Abuf[cur ^ 1]), Abase + ((s + 1) << 5), K, K);
      tdm_load_tile(lds_off_u32(Bbuf[cur ^ 1]), Wbase + ((s + 1) << 5), K, K);
    }
    const bf16x16 a0 = lds_frag_a(Abuf[cur], wm);
    const bf16x16 a1 = lds_frag_a(Abuf[cur], wm + 16);
    const bf16x16 b0 = lds_frag_b(Bbuf[cur], wn);
    const bf16x16 b1 = lds_frag_b(Bbuf[cur], wn + 16);
    const bf16x16 b2 = lds_frag_b(Bbuf[cur], wn + 32);
    const bf16x16 b3 = lds_frag_b(Bbuf[cur], wn + 48);
    acc[0][0] = __builtin_amdgcn_wmma_f32_16x16x32_bf16(
        false, a0, false, b0, (short)0, acc[0][0], false, false);
    acc[1][0] = __builtin_amdgcn_wmma_f32_16x16x32_bf16(
        false, a1, false, b0, (short)0, acc[1][0], false, false);
    acc[0][1] = __builtin_amdgcn_wmma_f32_16x16x32_bf16(
        false, a0, false, b1, (short)0, acc[0][1], false, false);
    acc[1][1] = __builtin_amdgcn_wmma_f32_16x16x32_bf16(
        false, a1, false, b1, (short)0, acc[1][1], false, false);
    acc[0][2] = __builtin_amdgcn_wmma_f32_16x16x32_bf16(
        false, a0, false, b2, (short)0, acc[0][2], false, false);
    acc[1][2] = __builtin_amdgcn_wmma_f32_16x16x32_bf16(
        false, a1, false, b2, (short)0, acc[1][2], false, false);
    acc[0][3] = __builtin_amdgcn_wmma_f32_16x16x32_bf16(
        false, a0, false, b3, (short)0, acc[0][3], false, false);
    acc[1][3] = __builtin_amdgcn_wmma_f32_16x16x32_bf16(
        false, a1, false, b3, (short)0, acc[1][3], false, false);
  }

  // C/D layout: VGPR r -> M = base + r + 8*(lane>=16); N = base + (lane&15).
  const int n = lane & 15;
  const int mrow = (lane >> 4) << 3;
#pragma unroll
  for (int tm = 0; tm < 2; ++tm) {
#pragma unroll
    for (int tn = 0; tn < 4; ++tn) {
      const size_t crow = (size_t)(blockM + wm + tm * 16 + mrow);
      const size_t ccol = (size_t)(blockN + wn + tn * 16 + n);
#pragma unroll
      for (int r = 0; r < 8; ++r) {
        float v = acc[tm][tn][r];
        if (MODE == 1) {
          v = v > 0.f ? v : 0.f;
          Cb[(crow + r) * (size_t)N + ccol] = f32_to_bf16_rne(v);
        } else {
          Cf[(crow + r) * (size_t)N + ccol] = v;
        }
      }
    }
  }
}

__global__ void cvt_f32_to_bf16(const float* __restrict__ in,
                                unsigned short* __restrict__ out, int n4) {
  int i = blockIdx.x * blockDim.x + threadIdx.x;
  if (i >= n4) return;
  float4 v = ((const float4*)in)[i];
  ushort4 o;
  o.x = f32_to_bf16_rne(v.x);
  o.y = f32_to_bf16_rne(v.y);
  o.z = f32_to_bf16_rne(v.z);
  o.w = f32_to_bf16_rne(v.w);
  ((ushort4*)out)[i] = o;
}

__global__ void zero_f32(float* __restrict__ p, int n) {
  int i = blockIdx.x * blockDim.x + threadIdx.x;
  if (i < n) p[i] = 0.f;
}

// h[b,j] = |proj[t,b,j] + HH[j]*h[b,j]| iterated over TC_ timesteps.
__global__ void recur_chunk(const float* __restrict__ proj,
                            const float* __restrict__ HH,
                            float* __restrict__ h) {
  int idx = blockIdx.x * blockDim.x + threadIdx.x;  // < B_*H_
  int j = idx & (H_ - 1);
  float d = HH[j];
  float hv = h[idx];
  const float* p = proj + idx;
#pragma unroll 4
  for (int t = 0; t < TC_; ++t) {
    hv = fabsf(*p + d * hv);
    p += B_ * H_;
  }
  h[idx] = hv;
}

// out[b,o] = dot(h[b,:], W_ho[o,:]) + b_ho[o]   (tiny; fp32)
__global__ void out_proj_kernel(const float* __restrict__ h,
                                const float* __restrict__ Who,
                                const float* __restrict__ bho,
                                float* __restrict__ out) {
  int idx = blockIdx.x * blockDim.x + threadIdx.x;  // < B_*O_
  int o = idx & (O_ - 1);
  int b = idx >> 9;  // O_ = 512
  const float4* hp = (const float4*)(h + (size_t)b * H_);
  const float4* wp = (const float4*)(Who + (size_t)o * H_);
  float s = 0.f;
#pragma unroll 4
  for (int k = 0; k < H_ / 4; ++k) {
    float4 a = hp[k], w = wp[k];
    s += a.x * w.x + a.y * w.y + a.z * w.z + a.w * w.w;
  }
  out[idx] = s + bho[o];
}

extern "C" void kernel_launch(void* const* d_in, const int* in_sizes, int n_in,
                              void* d_out, int out_size, void* d_ws,
                              size_t ws_size, hipStream_t stream) {
  (void)in_sizes; (void)n_in; (void)out_size; (void)ws_size;

  const float* X      = (const float*)d_in[0];  // [T,B,I]
  const float* W_ih   = (const float*)d_in[1];  // [G,I]
  const float* W_cell = (const float*)d_in[2];  // [H,G]
  const float* HH     = (const float*)d_in[3];  // [H]
  const float* W_ho   = (const float*)d_in[4];  // [O,H]
  const float* b_ho   = (const float*)d_in[5];  // [O]
  float* out = (float*)d_out;                   // [B,O]

  char* w = (char*)d_ws;
  auto take = [&](size_t bytes) -> char* {
    char* p = w;
    w += (bytes + 255) & ~(size_t)255;
    return p;
  };
  bf16_t* wih_bf   = (bf16_t*)take((size_t)G_ * I_ * 2);   //  1 MiB
  bf16_t* wcell_bf = (bf16_t*)take((size_t)H_ * G_ * 2);   //  2 MiB
  bf16_t* x_bf     = (bf16_t*)take((size_t)MC_ * I_ * 2);  //  8 MiB
  bf16_t* g_bf     = (bf16_t*)take((size_t)MC_ * G_ * 2);  // 16 MiB
  float*  proj     = (float*) take((size_t)MC_ * H_ * 4);  // 32 MiB
  float*  hbuf     = (float*) take((size_t)B_ * H_ * 4);   // 0.5 MiB

  // One-time weight conversion + h init.
  cvt_f32_to_bf16<<<(G_ * I_ / 4 + 255) / 256, 256, 0, stream>>>(
      W_ih, (unsigned short*)wih_bf, G_ * I_ / 4);
  cvt_f32_to_bf16<<<(H_ * G_ / 4 + 255) / 256, 256, 0, stream>>>(
      W_cell, (unsigned short*)wcell_bf, H_ * G_ / 4);
  zero_f32<<<(B_ * H_) / 256, 256, 0, stream>>>(hbuf, B_ * H_);

  const dim3 grid1(G_ / 128, MC_ / 128);
  const dim3 grid2(H_ / 128, MC_ / 128);
  for (int c = 0; c < T_ / TC_; ++c) {
    const float* Xc = X + (size_t)c * TC_ * B_ * I_;
    cvt_f32_to_bf16<<<(MC_ * I_ / 4) / 256, 256, 0, stream>>>(
        Xc, (unsigned short*)x_bf, MC_ * I_ / 4);
    // g = relu(Xc @ W_ih^T), stored bf16
    wmma_gemm_tdm<1><<<grid1, 256, 0, stream>>>(
        x_bf, wih_bf, nullptr, (unsigned short*)g_bf, MC_, G_, I_);
    // proj = g @ W_cell^T, stored fp32
    wmma_gemm_tdm<0><<<grid2, 256, 0, stream>>>(
        g_bf, wcell_bf, proj, nullptr, MC_, H_, G_);
    // sequential diagonal recurrence over this chunk's timesteps
    recur_chunk<<<(B_ * H_) / 256, 256, 0, stream>>>(proj, HH, hbuf);
  }

  out_proj_kernel<<<(B_ * O_) / 256, 256, 0, stream>>>(hbuf, W_ho, b_ho, out);
}
